// MultiHeadAttention_81372450390745
// MI455X (gfx1250) — matrix operand order repaired and checked by reading, not compiled, not run on hardware
//
#include <hip/hip_runtime.h>

typedef __attribute__((ext_vector_type(16))) _Float16 v16h;
typedef __attribute__((ext_vector_type(8)))  float    v8f;

#define Bq 4
#define Sq 2048
#define Dq 1024
#define Hq 16
#define HDq 64

// ---------------------------------------------------------------------------
// CDNA5 async global->LDS copy (ASYNCcnt-tracked, no VGPR round trip).
__device__ __forceinline__ void async_ld_b128(uint32_t lds_off, const void* g) {
  asm volatile("global_load_async_to_lds_b128 %0, %1, off"
               :: "v"(lds_off), "v"(g) : "memory");
}
__device__ __forceinline__ void wait_asynccnt0() {
  asm volatile("s_wait_asynccnt 0" ::: "memory");
}
// allow N outstanding (the just-issued next tile); older ones complete in order
__device__ __forceinline__ void wait_asynccnt8() {
  asm volatile("s_wait_asynccnt 8" ::: "memory");
}
__device__ __forceinline__ void wait_asynccnt2() {
  asm volatile("s_wait_asynccnt 2" ::: "memory");
}
__device__ __forceinline__ void wait_dscnt0() {
  asm volatile("s_wait_dscnt 0" ::: "memory");
}

// ---------------------------------------------------------------------------
// DPP16 butterfly reductions over 16-lane groups (pure VALU, no ds_bpermute)
template <int C>
__device__ __forceinline__ float dpp_mov(float x) {
  return __int_as_float(
      __builtin_amdgcn_update_dpp(0, __float_as_int(x), C, 0xF, 0xF, false));
}
__device__ __forceinline__ float red_max16(float x) {
  x = fmaxf(x, dpp_mov<0xB1>(x));   // quad_perm(1,0,3,2)  xor 1
  x = fmaxf(x, dpp_mov<0x4E>(x));   // quad_perm(2,3,0,1)  xor 2
  x = fmaxf(x, dpp_mov<0x141>(x));  // row_half_mirror     cross 4
  x = fmaxf(x, dpp_mov<0x140>(x));  // row_mirror          cross 8
  return x;
}
__device__ __forceinline__ float red_sum16(float x) {
  x += dpp_mov<0xB1>(x);
  x += dpp_mov<0x4E>(x);
  x += dpp_mov<0x141>(x);
  x += dpp_mov<0x140>(x);
  return x;
}

// ---------------------------------------------------------------------------
// Fragment loaders (CDNA5 16-bit WMMA VGPR layouts, wave32)
__device__ __forceinline__ v16h load_a_frag(const _Float16* base, int rowBase,
                                            int stride, int kBase, int lane) {
  union { v16h v; float4 f[2]; } u;
  const _Float16* p = base + (size_t)(rowBase + (lane & 15)) * stride + kBase +
                      ((lane >> 4) << 3);
  u.f[0] = *reinterpret_cast<const float4*>(p);
  u.f[1] = *reinterpret_cast<const float4*>(p + 16);
  return u.v;
}
__device__ __forceinline__ v16h load_b_frag(const _Float16* base, int colBase,
                                            int stride, int kBase, int lane) {
  union { v16h v; float4 f[2]; } u;
  const _Float16* p = base + (size_t)(colBase + (lane & 15)) * stride + kBase +
                      ((lane >> 4) << 3);
  u.f[0] = *reinterpret_cast<const float4*>(p);
  u.f[1] = *reinterpret_cast<const float4*>(p + 16);
  return u.v;
}

__device__ __forceinline__ v8f wmma_f16(v16h a, v16h b, v8f c) {
  return __builtin_amdgcn_wmma_f32_16x16x32_f16(false, a, false, b, (short)0, c,
                                                false, false);
}

// ---------------------------------------------------------------------------
__global__ __launch_bounds__(256) void cvt16_kernel(const float* __restrict__ s,
                                                    _Float16* __restrict__ d) {
  const int i = blockIdx.x * blockDim.x + threadIdx.x;
  float4 f = reinterpret_cast<const float4*>(s)[i];
  union { _Float16 h[4]; float2 f2; } u;
  u.h[0] = (_Float16)f.x; u.h[1] = (_Float16)f.y;
  u.h[2] = (_Float16)f.z; u.h[3] = (_Float16)f.w;
  reinterpret_cast<float2*>(d)[i] = u.f2;
}

__global__ __launch_bounds__(256) void transpose_cvt_kernel(
    const float* __restrict__ W, _Float16* __restrict__ WT) {
  __shared__ float T[32][33];
  const int tx = threadIdx.x & 31;
  const int ty0 = threadIdx.x >> 5;
  const int nt = blockIdx.x * 32, kt = blockIdx.y * 32;
#pragma unroll
  for (int r = 0; r < 4; ++r) {
    const int ky = ty0 + r * 8;
    T[ky][tx] = W[(size_t)(kt + ky) * Dq + nt + tx];
  }
  __syncthreads();
#pragma unroll
  for (int r = 0; r < 4; ++r) {
    const int ny = ty0 + r * 8;
    WT[(size_t)(nt + ny) * Dq + kt + tx] = (_Float16)T[tx][ny];
  }
}

// ---------------------------------------------------------------------------
// GEMM: C(8192x1024) = A16(8192x1024) * BT16([n][k]).
// BK=64, double-buffered async LDS pipeline, 16 WMMAs per stage per wave.
template <int MODE>
__global__ __launch_bounds__(256) void gemm16_kernel(
    const _Float16* __restrict__ A, const _Float16* __restrict__ BT,
    float* __restrict__ dst, _Float16* __restrict__ dst16) {
  __shared__ _Float16 Ash[2][128][72];  // [m][k64], 144B rows (128B + pad)
  __shared__ _Float16 Bsh[2][128][72];  // [n][k64]

  const int tid = threadIdx.x;
  const int lane = tid & 31;
  const int wid = tid >> 5;
  const int wm = wid >> 1, wn = wid & 1;
  const int mBase = blockIdx.y * 128;
  const int nBase = blockIdx.x * 128;

  auto stage = [&](int buf, int kb) {
    const uint32_t aB = (uint32_t)(uintptr_t)&Ash[buf][0][0];
    const uint32_t bB = (uint32_t)(uintptr_t)&Bsh[buf][0][0];
#pragma unroll
    for (int r = 0; r < 4; ++r) {
      const uint32_t o = (uint32_t)tid * 16u + (uint32_t)r * 4096u;
      const uint32_t row = o >> 7, colB = o & 127u;
      async_ld_b128(aB + row * 144u + colB,
                    (const char*)A + ((size_t)(mBase + row) * Dq + kb) * 2 + colB);
      async_ld_b128(bB + row * 144u + colB,
                    (const char*)BT + ((size_t)(nBase + row) * Dq + kb) * 2 + colB);
    }
  };

  v8f c[2][4];
  const v8f vzero = {};
#pragma unroll
  for (int i = 0; i < 2; ++i)
#pragma unroll
    for (int j = 0; j < 4; ++j) c[i][j] = vzero;

  stage(0, 0);
  int buf = 0;
  for (int kb = 0; kb < Dq; kb += 64, buf ^= 1) {
    __syncthreads();  // all waves done reading buf^1 (previous stage)
    if (kb + 64 < Dq) {
      stage(buf ^ 1, kb + 64);  // prefetch next tile under compute
      wait_asynccnt8();         // current tile resident (in-order completion)
    } else {
      wait_asynccnt0();
    }
    __syncthreads();  // every wave's current tile resident

    const _Float16* Ab = &Ash[buf][0][0];
    const _Float16* Bb = &Bsh[buf][0][0];
#pragma unroll
    for (int kc = 0; kc < 2; ++kc) {
      v16h a0 = load_a_frag(Ab, wm * 32, 72, kc * 32, lane);
      v16h a1 = load_a_frag(Ab, wm * 32 + 16, 72, kc * 32, lane);
      v16h b[4];
#pragma unroll
      for (int j = 0; j < 4; ++j)
        b[j] = load_b_frag(Bb, wn * 64 + j * 16, 72, kc * 32, lane);
#pragma unroll
      for (int j = 0; j < 4; ++j) {
        c[0][j] = wmma_f16(a0, b[j], c[0][j]);
        c[1][j] = wmma_f16(a1, b[j], c[1][j]);
      }
    }
  }

#pragma unroll
  for (int i = 0; i < 2; ++i) {
#pragma unroll
    for (int j = 0; j < 4; ++j) {
      const int col = nBase + wn * 64 + j * 16 + (lane & 15);
#pragma unroll
      for (int vv = 0; vv < 8; ++vv) {
        const int row = mBase + wm * 32 + i * 16 + vv + ((lane >> 4) << 3);
        const float val = c[i][j][vv];
        if constexpr (MODE == 0) {
          dst[(size_t)row * Dq + col] = val;
        } else {
          const int bb = row >> 11, ss = row & 2047;
          const int hh = col >> 6, dd = col & 63;
          const size_t idx = (((size_t)(bb * Hq + hh) * Sq) + ss) * HDq + dd;
          dst[idx] = val;
          if constexpr (MODE == 2) dst16[idx] = (_Float16)val;
        }
      }
    }
  }
}

// ---------------------------------------------------------------------------
// RoPE on (B,H,S,hd) f32; writes f16 copy (SCALE folds softmax 1/64 into q).
template <bool WRITE_F32, bool SCALE>
__global__ __launch_bounds__(256) void rope_kernel(float* __restrict__ t,
                                                   _Float16* __restrict__ o16) {
  const int idx = blockIdx.x * blockDim.x + threadIdx.x;  // 2^22
  const int i = idx & 31;
  const int s = (idx >> 5) & (Sq - 1);
  const int bh = idx >> 16;
  const float inv_freq = __expf(-(float)(2 * i) * (1.0f / 64.0f) * 9.210340372f);
  const float ang = (float)s * inv_freq;
  float sn, cs;
  __sincosf(ang, &sn, &cs);
  const size_t base = ((size_t)bh * Sq + s) * HDq + i;
  const float x1 = t[base];
  const float x2 = t[base + 32];
  const float r1 = x1 * cs - x2 * sn;
  const float r2 = x2 * cs + x1 * sn;
  if constexpr (WRITE_F32) { t[base] = r1; t[base + 32] = r2; }
  const float sc = SCALE ? (1.0f / 64.0f) : 1.0f;
  o16[base] = (_Float16)(r1 * sc);
  o16[base + 32] = (_Float16)(r2 * sc);
}

// ---------------------------------------------------------------------------
// Flash attention: block per (b,h,qtile64), 4 waves x 16 queries.
// Double-buffered async K tiles; DPP16 softmax reductions.
__global__ __launch_bounds__(128) void attn_wmma_kernel(
    const _Float16* __restrict__ q16, const _Float16* __restrict__ k16,
    const _Float16* __restrict__ v16, _Float16* __restrict__ ctx16) {
  __shared__ _Float16 Qsh[64][72];
  __shared__ _Float16 Ksh[2][32][72];
  __shared__ _Float16 Vts[64][40];
  __shared__ _Float16 Psh[4][16][40];

  const int tid = threadIdx.x;
  const int lane = tid & 31;
  const int wid = tid >> 5;
  const int qt = blockIdx.x & 31;
  const int bh = blockIdx.x >> 5;
  const int bb = bh >> 4, hh = bh & 15;
  const int qb0 = qt * 64;

  const _Float16* qp = q16 + (size_t)bh * Sq * HDq;
  const _Float16* kp = k16 + (size_t)bh * Sq * HDq;
  const _Float16* vp = v16 + (size_t)bh * Sq * HDq;

  auto stage_k = [&](int buf, int kb) {
    const uint32_t kB = (uint32_t)(uintptr_t)&Ksh[buf][0][0];
#pragma unroll
    for (int r = 0; r < 2; ++r) {
      const uint32_t o = (uint32_t)tid * 16u + (uint32_t)r * 2048u;
      const uint32_t key = o >> 7, colB = o & 127u;
      async_ld_b128(kB + key * 144u + colB,
                    (const char*)kp + ((size_t)(kb + key) * HDq) * 2 + colB);
    }
  };

  // ---- async-stage Q tile (64 x 128B rows) + first K tile
  {
    const uint32_t qB = (uint32_t)(uintptr_t)&Qsh[0][0];
#pragma unroll
    for (int r = 0; r < 4; ++r) {
      const uint32_t o = (uint32_t)tid * 16u + (uint32_t)r * 2048u;
      const uint32_t row = o >> 7, colB = o & 127u;
      async_ld_b128(qB + row * 144u + colB,
                    (const char*)qp + ((size_t)(qb0 + row) * HDq) * 2 + colB);
    }
  }
  stage_k(0, 0);
  wait_asynccnt0();
  __syncthreads();

  const v16h aq0 = load_a_frag(&Qsh[0][0], wid * 16, 72, 0, lane);
  const v16h aq1 = load_a_frag(&Qsh[0][0], wid * 16, 72, 32, lane);

  float m[8], l[8];
  v8f o[4];
  const v8f vzero = {};
#pragma unroll
  for (int vv = 0; vv < 8; ++vv) { m[vv] = -1e30f; l[vv] = 0.0f; }
#pragma unroll
  for (int j = 0; j < 4; ++j) o[j] = vzero;

  const int nkt = (qb0 + 64) >> 5;
  int buf = 0;
  for (int kt = 0; kt < nkt; ++kt, buf ^= 1) {
    const int kb = kt * 32;
    __syncthreads();  // prev reads of Vts / Ksh[buf^1] complete
    // ---- stage V^T ([hd][key], f16 transpose, ds stores)
    {
      const int j = (tid & 15) * 4;
      const int i0 = tid >> 4;
#pragma unroll
      for (int r = 0; r < 4; ++r) {
        const int key = i0 + r * 8;
        union { uint2 u2; _Float16 h[4]; } g;
        g.u2 = *reinterpret_cast<const uint2*>(vp + (size_t)(kb + key) * HDq + j);
        Vts[j + 0][key] = g.h[0];
        Vts[j + 1][key] = g.h[1];
        Vts[j + 2][key] = g.h[2];
        Vts[j + 3][key] = g.h[3];
      }
    }
    if (kt + 1 < nkt) {
      stage_k(buf ^ 1, kb + 32);  // prefetch next K tile
      wait_asynccnt2();           // current K tile resident
    } else {
      wait_asynccnt0();
    }
    __syncthreads();

    // ---- scores (q pre-scaled by 1/64)
    const _Float16* Kb = &Ksh[buf][0][0];
    v8f s[2];
#pragma unroll
    for (int g = 0; g < 2; ++g) {
      v16h bk0 = load_b_frag(Kb, g * 16, 72, 0, lane);
      v16h bk1 = load_b_frag(Kb, g * 16, 72, 32, lane);
      v8f sc = vzero;
      sc = wmma_f16(aq0, bk0, sc);
      sc = wmma_f16(aq1, bk1, sc);
      s[g] = sc;
    }

    // ---- causal mask
    const int rowB = qb0 + wid * 16 + ((lane >> 4) << 3);
#pragma unroll
    for (int g = 0; g < 2; ++g) {
      const int col = kb + g * 16 + (lane & 15);
#pragma unroll
      for (int vv = 0; vv < 8; ++vv)
        if (col > rowB + vv) s[g][vv] = -1e30f;
    }

    // ---- online softmax with DPP16 reductions
    float rs[8];
#pragma unroll
    for (int vv = 0; vv < 8; ++vv) {
      const float mx = red_max16(fmaxf(s[0][vv], s[1][vv]));
      const float mnew = fmaxf(m[vv], mx);
      const float sf = __expf(m[vv] - mnew);
      m[vv] = mnew;
      l[vv] *= sf;
#pragma unroll
      for (int j = 0; j < 4; ++j) o[j][vv] *= sf;
      const float p0 = __expf(s[0][vv] - mnew);
      const float p1 = __expf(s[1][vv] - mnew);
      const int prow = vv + ((lane >> 4) << 3);
      Psh[wid][prow][lane & 15] = (_Float16)p0;
      Psh[wid][prow][16 + (lane & 15)] = (_Float16)p1;
      rs[vv] = p0 + p1;
    }
#pragma unroll
    for (int vv = 0; vv < 8; ++vv) l[vv] += red_sum16(rs[vv]);

    // same-wave LDS round trip for P (C-layout -> A-layout)
    wait_dscnt0();
    const v16h ap = load_a_frag(&Psh[wid][0][0], 0, 40, 0, lane);
#pragma unroll
    for (int j = 0; j < 4; ++j) {
      v16h bv = load_b_frag(&Vts[0][0], j * 16, 40, 0, lane);
      o[j] = wmma_f16(ap, bv, o[j]);
    }
  }

  // ---- normalize and write ctx16 in (B,S,D)
#pragma unroll
  for (int vv = 0; vv < 8; ++vv) {
    const float inv = 1.0f / l[vv];
    const int srow = qb0 + wid * 16 + vv + ((lane >> 4) << 3);
#pragma unroll
    for (int j = 0; j < 4; ++j) {
      const int col = hh * HDq + j * 16 + (lane & 15);
      ctx16[((size_t)(bb * Sq + srow)) * Dq + col] = (_Float16)(o[j][vv] * inv);
    }
  }
}

// ---------------------------------------------------------------------------
extern "C" void kernel_launch(void* const* d_in, const int* in_sizes, int n_in,
                              void* d_out, int out_size, void* d_ws,
                              size_t ws_size, hipStream_t stream) {
  const float* x  = (const float*)d_in[0];
  const float* Wq = (const float*)d_in[1];
  const float* Wk = (const float*)d_in[2];
  const float* Wv = (const float*)d_in[3];
  const float* Wo = (const float*)d_in[4];

  const size_t NBD = (size_t)Bq * Sq * Dq;  // 8388608
  const size_t NW  = (size_t)Dq * Dq;       // 1048576
  float* out  = (float*)d_out;
  float* kout = out + NBD;
  float* vout = out + 2 * NBD;

  _Float16* x16 = (_Float16*)d_ws;
  _Float16* wqt = x16 + NBD;
  _Float16* wkt = wqt + NW;
  _Float16* wvt = wkt + NW;
  _Float16* wot = wvt + NW;
  _Float16* q16 = wot + NW;
  _Float16* k16 = q16 + NBD;
  _Float16* v16 = k16 + NBD;
  float*    q32 = (float*)(v16 + NBD);
  _Float16* ctx16 = x16;  // x16 dead after the QKV GEMMs -> reuse

  cvt16_kernel<<<NBD / 4 / 256, 256, 0, stream>>>(x, x16);
  dim3 tgrid(32, 32);
  transpose_cvt_kernel<<<tgrid, 256, 0, stream>>>(Wq, wqt);
  transpose_cvt_kernel<<<tgrid, 256, 0, stream>>>(Wk, wkt);
  transpose_cvt_kernel<<<tgrid, 256, 0, stream>>>(Wv, wvt);
  transpose_cvt_kernel<<<tgrid, 256, 0, stream>>>(Wo, wot);

  dim3 ggrid(Dq / 128, (Bq * Sq) / 128);  // 8 x 64
  gemm16_kernel<1><<<ggrid, 256, 0, stream>>>(x16, wqt, q32, nullptr);
  gemm16_kernel<1><<<ggrid, 256, 0, stream>>>(x16, wkt, kout, nullptr);
  gemm16_kernel<2><<<ggrid, 256, 0, stream>>>(x16, wvt, vout, v16);

  const int ropeN = Bq * Hq * Sq * 32;  // 2^22
  rope_kernel<false, true><<<ropeN / 256, 256, 0, stream>>>(q32, q16);
  rope_kernel<true, false><<<ropeN / 256, 256, 0, stream>>>(kout, k16);

  attn_wmma_kernel<<<Bq * Hq * (Sq / 64), 128, 0, stream>>>(q16, k16, v16, ctx16);

  gemm16_kernel<0><<<ggrid, 256, 0, stream>>>(ctx16, wot, out, nullptr);
}
